// GnnMapping_75282186764487
// MI455X (gfx1250) — compile-verified, matrix operand order checked by reading, not run on hardware
//
#include <hip/hip_runtime.h>
#include <cstdint>

typedef __attribute__((ext_vector_type(16))) __bf16 v16bf;
typedef __attribute__((ext_vector_type(8)))  float  v8f;
typedef unsigned short ushort_t;

union FragB {
    v16bf v;
    ushort_t u[16];
    uint4 q[2];
};

__device__ __forceinline__ ushort_t f2bf(float x) {
    unsigned int u = __float_as_uint(x);
    unsigned int r = u + 0x7FFFu + ((u >> 16) & 1u);   // round-to-nearest-even
    return (ushort_t)(r >> 16);
}

__device__ __forceinline__ float lrelu(float v, float s) { return v > 0.f ? v : s * v; }

__device__ __forceinline__ void atomicMaxF(float* addr, float val) {
    int* ia = (int*)addr;
    int old = *ia;
    while (__int_as_float(old) < val) {
        int assumed = old;
        old = atomicCAS(ia, assumed, __float_as_int(val));
        if (old == assumed) break;
    }
}

// ---------------------------------------------------------------------------
// Pack fp32 weight W[K,Nc] into bf16 WMMA B-fragment order:
//   Wp[((kt*nTiles + nt)*32 + lane)*16 + j]
// with (lane,j) -> (k,n):  n = lane&15,
//   i=j>>1,  k = (i>>2)*16 + (lane>>4)*8 + (i&3)*2 + (j&1)
// so each lane's 16 bf16 (32B) are contiguous -> two global_load_b128.
// ---------------------------------------------------------------------------
__global__ void k_pack_w(const float* __restrict__ W, ushort_t* __restrict__ Wp,
                         int K, int Nc) {
    int t = blockIdx.x * blockDim.x + threadIdx.x;
    int total = K * Nc;
    if (t >= total) return;
    int nTiles = Nc >> 4;
    int j    = t & 15;
    int lane = (t >> 4) & 31;
    int rem  = t >> 9;
    int nt   = rem % nTiles;
    int kt   = rem / nTiles;
    int i    = j >> 1;
    int nloc = lane & 15;
    int k = ((i >> 2) << 4) + ((lane >> 4) << 3) + ((i & 3) << 1) + (j & 1);
    int kg = (kt << 5) + k;
    int n  = (nt << 4) + nloc;
    Wp[t] = f2bf(W[(size_t)kg * Nc + n]);
}

// ---------------------------------------------------------------------------
// GEMM: C[M,Nc] = act( A[M,K](f32) @ W(bf16 packed) + bias )
// block = 128 threads = 4 waves; each wave owns NT 16x16 output tiles,
// grid.x * 4 * NT == Nc/16 exactly -> no guards, WMMAs are unconditional
// straight-line code (accumulators stay pinned, no phi copies).
// A 16x32 fp32 tile staged into LDS by GLOBAL_LOAD_ASYNC_TO_LDS_B128,
// double-buffered so the DMA of tile kt+1 overlaps the WMMAs of tile kt
// (ASYNCcnt + barrier sync). Per-lane fragment build converts f32->bf16
// from LDS. Row stride padded to 36 floats: 16B-aligned b128 LDS writes,
// 8B-aligned float2 reads, conflict-free bank stride.
// ---------------------------------------------------------------------------
#define A_STRIDE 36

template <int NT>
__global__ __launch_bounds__(128)
void k_gemm_bf16(const float* __restrict__ A, const ushort_t* __restrict__ Wp,
                 const float* __restrict__ bias, float* __restrict__ C,
                 int Nc, int K, int act) {
    __shared__ __align__(16) float aLdsF[2][16 * A_STRIDE];
    const int tid    = threadIdx.x;
    const int lane   = tid & 31;
    const int wave   = tid >> 5;
    const int mtile  = blockIdx.y;
    const int nTiles = Nc >> 4;
    const int nt0    = blockIdx.x * (4 * NT) + wave * NT;
    const int kTiles = K >> 5;

    // async staging geometry: 128 threads x 16B = 2KB = 16x32 f32 tile
    const int row = tid >> 3;       // 0..15
    const int seg = tid & 7;        // 0..7 (4 floats each)
    const float* gRow = A + (size_t)(mtile * 16 + row) * K + seg * 4;
    unsigned int ldsOff0 = (unsigned int)(uintptr_t)&aLdsF[0][row * A_STRIDE + seg * 4];
    unsigned int ldsOff1 = (unsigned int)(uintptr_t)&aLdsF[1][row * A_STRIDE + seg * 4];

    // prologue: kick off K-tile 0
    {
        unsigned long long ga = (unsigned long long)(uintptr_t)gRow;
        asm volatile("global_load_async_to_lds_b128 %0, %1, off"
                     :: "v"(ldsOff0), "v"(ga) : "memory");
    }

    v8f acc[NT];
#pragma unroll
    for (int qq = 0; qq < NT; ++qq) acc[qq] = v8f{};

    for (int kt = 0; kt < kTiles; ++kt) {
        __syncthreads();   // all waves done reading buffer (kt+1)&1 (from iter kt-1)
        if (kt + 1 < kTiles) {
            unsigned long long ga =
                (unsigned long long)(uintptr_t)(gRow + (size_t)(kt + 1) * 32);
            unsigned int dstOff = ((kt + 1) & 1) ? ldsOff1 : ldsOff0;
            asm volatile("global_load_async_to_lds_b128 %0, %1, off"
                         :: "v"(dstOff), "v"(ga) : "memory");
            asm volatile("s_wait_asynccnt 1" ::: "memory");   // tile kt landed
        } else {
            asm volatile("s_wait_asynccnt 0" ::: "memory");
        }
        __syncthreads();   // tile kt visible to every wave

        // build A fragment: lane m = lane&15, element pair i ->
        //   k = (i>>2)*16 + (lane>>4)*8 + (i&3)*2
        const float* rowp = &aLdsF[kt & 1][(lane & 15) * A_STRIDE];
        const int khalf = (lane >> 4) << 3;
        FragB af;
#pragma unroll
        for (int i = 0; i < 8; ++i) {
            int k0 = ((i >> 2) << 4) + khalf + ((i & 3) << 1);
            float2 f2 = *(const float2*)(rowp + k0);
            af.u[2 * i]     = f2bf(f2.x);
            af.u[2 * i + 1] = f2bf(f2.y);
        }

        if (kt + 1 < kTiles)   // prefetch next K-tile of B (global_prefetch_b8)
            __builtin_prefetch(Wp + (((size_t)(kt + 1) * nTiles + nt0) * 32 + lane) * 16, 0, 1);

        const ushort_t* wBase = Wp + (((size_t)kt * nTiles + nt0) * 32 + lane) * 16;
#pragma unroll
        for (int qq = 0; qq < NT; ++qq) {
            FragB bf;
            const uint4* wp = (const uint4*)(wBase + (size_t)qq * 512);
            bf.q[0] = wp[0];
            bf.q[1] = wp[1];
            acc[qq] = __builtin_amdgcn_wmma_f32_16x16x32_bf16(
                false, af.v, false, bf.v, (short)0, acc[qq], false, false);
        }
    }

    // C/D layout: lane n=lane&15, rows r + 8*(lane>>4)
    int nloc  = lane & 15;
    int mbase = (lane >> 4) << 3;
#pragma unroll
    for (int qq = 0; qq < NT; ++qq) {
        int ncol = ((nt0 + qq) << 4) + nloc;
        float bv = bias ? bias[ncol] : 0.f;
#pragma unroll
        for (int r = 0; r < 8; ++r) {
            int mrow = mtile * 16 + mbase + r;
            float v = acc[qq][r] + bv;
            if (act) v = lrelu(v, 0.01f);
            C[(size_t)mrow * Nc + ncol] = v;
        }
    }
}

// ---------------------------------------------------------------------------
// Edge / segment kernels
// ---------------------------------------------------------------------------
__global__ void k_build_edges(const int* __restrict__ ei, int* __restrict__ src2,
                              int* __restrict__ dst2, int E, int Nn) {
    int t = blockIdx.x * blockDim.x + threadIdx.x;
    if (t < E) { src2[t] = ei[t]; dst2[t] = ei[E + t]; }
    else if (t < E + Nn) { src2[t] = t - E; dst2[t] = t - E; }  // self loops
}

__global__ void k_fill(float* __restrict__ p, float v, int n) {
    int t = blockIdx.x * blockDim.x + threadIdx.x;
    if (t < n) p[t] = v;
}

__global__ __launch_bounds__(256)
void k_edge_logit(const float* __restrict__ xl, const float* __restrict__ xr,
                  const float* __restrict__ att, const int* __restrict__ src,
                  const int* __restrict__ dst, float* __restrict__ logits,
                  float* __restrict__ mbuf, int E2, int H, int Nout) {
    int w    = (blockIdx.x * blockDim.x + threadIdx.x) >> 5;   // one wave per (edge,head)
    int lane = threadIdx.x & 31;
    int e = w / H, h = w - e * H;
    if (e >= E2) return;
    int s = src[e], d = dst[e];
    const float* pl = xl + (size_t)s * Nout + h * 512;
    const float* pr = xr + (size_t)d * Nout + h * 512;
    const float* pa = att + h * 512;
    float acc = 0.f;
#pragma unroll 4
    for (int c = lane; c < 512; c += 32) {
        float v = pl[c] + pr[c];
        acc += pa[c] * lrelu(v, 0.2f);
    }
    for (int off = 16; off > 0; off >>= 1) acc += __shfl_down(acc, off, 32);
    if (lane == 0) {
        logits[(size_t)e * H + h] = acc;
        atomicMaxF(mbuf + (size_t)d * H + h, acc);
    }
}

__global__ void k_edge_alpha(const float* __restrict__ logits, const float* __restrict__ mbuf,
                             const int* __restrict__ dst, float* __restrict__ pbuf,
                             float* __restrict__ den, int E2, int H) {
    int t = blockIdx.x * blockDim.x + threadIdx.x;
    if (t >= E2 * H) return;
    int e = t / H, h = t - e * H;
    float p = __expf(logits[t] - mbuf[dst[e] * H + h]);
    pbuf[t] = p;
    atomicAdd(den + dst[e] * H + h, p);
}

__global__ __launch_bounds__(256)
void k_edge_scatter(const float* __restrict__ xl, const float* __restrict__ pbuf,
                    const float* __restrict__ den, const int* __restrict__ src,
                    const int* __restrict__ dst, float* __restrict__ outb,
                    int H, int Nout) {
    int e = blockIdx.x;
    int s = src[e], d = dst[e];
    for (int c = threadIdx.x; c < Nout; c += 256) {
        int h = c >> 9;                       // C = 512 per head
        float alpha = pbuf[(size_t)e * H + h] / den[(size_t)d * H + h];
        atomicAdd(outb + (size_t)d * Nout + c, alpha * xl[(size_t)s * Nout + c]);
    }
}

__global__ void k_bias_lrelu(const float* __restrict__ in, const float* __restrict__ bias,
                             float* __restrict__ out, int n, int Nout) {
    int t = blockIdx.x * blockDim.x + threadIdx.x;
    if (t >= n) return;
    out[t] = lrelu(in[t] + bias[t % Nout], 0.01f);
}

__global__ void k_add(float* __restrict__ a, const float* __restrict__ b, int n) {
    int t = blockIdx.x * blockDim.x + threadIdx.x;
    if (t < n) a[t] += b[t];
}

// PyG LayerNorm(mode='graph'): normalize over nodes AND channels per graph.
__global__ __launch_bounds__(256)
void k_layernorm(float* __restrict__ h, const float* __restrict__ lnw,
                 const float* __restrict__ lnb, int npg) {
    __shared__ float red[256];
    int g = blockIdx.x;
    int node = g * npg + threadIdx.x;
    float* row = h + (size_t)node * 512;
    float s = 0.f;
    for (int c = 0; c < 512; ++c) s += row[c];
    red[threadIdx.x] = s; __syncthreads();
    for (int off = 128; off > 0; off >>= 1) {
        if (threadIdx.x < off) red[threadIdx.x] += red[threadIdx.x + off];
        __syncthreads();
    }
    float mean = red[0] / (npg * 512.f);
    __syncthreads();
    float s2 = 0.f;
    for (int c = 0; c < 512; ++c) { float d = row[c] - mean; s2 += d * d; }
    red[threadIdx.x] = s2; __syncthreads();
    for (int off = 128; off > 0; off >>= 1) {
        if (threadIdx.x < off) red[threadIdx.x] += red[threadIdx.x + off];
        __syncthreads();
    }
    float var = red[0] / (npg * 512.f);
    float rs = rsqrtf(var + 1e-5f);
    for (int c = 0; c < 512; ++c)
        row[c] = (row[c] - mean) * rs * lnw[c] + lnb[c];
}

// SoftmaxAggregation(learn=True): channel-wise softmax over nodes per graph.
__global__ __launch_bounds__(256)
void k_aggr(const float* __restrict__ h, const float* __restrict__ tptr,
            float* __restrict__ g, int npg) {
    int gb = blockIdx.x;
    float t = *tptr;
    const float* base = h + (size_t)gb * npg * 512;
    for (int c = threadIdx.x; c < 512; c += 256) {
        float mx = -1e30f;
        for (int n = 0; n < npg; ++n) mx = fmaxf(mx, base[(size_t)n * 512 + c] * t);
        float ps = 0.f, ws = 0.f;
        for (int n = 0; n < npg; ++n) {
            float hv = base[(size_t)n * 512 + c];
            float p = __expf(hv * t - mx);
            ps += p; ws += p * hv;
        }
        g[gb * 512 + c] = ws / ps;
    }
}

// ---------------------------------------------------------------------------
extern "C" void kernel_launch(void* const* d_in, const int* in_sizes, int n_in,
                              void* d_out, int out_size, void* d_ws, size_t ws_size,
                              hipStream_t stream) {
    (void)n_in; (void)out_size; (void)ws_size;
    const int Nn = in_sizes[2];          // 16384 nodes
    const int Eo = in_sizes[1] / 2;      // 49152 edges
    const int Bn = 64;
    const int E2 = Eo + Nn;              // + self loops
    const int npg = Nn / Bn;             // 256 nodes/graph

    const float* x  = (const float*)d_in[0];
    const int*   ei = (const int*)d_in[1];

    // workspace carve-out
    char* base = (char*)d_ws;
    size_t off = 0;
    auto allocf = [&](size_t n) { float* p = (float*)(base + off); off += ((n * 4 + 255) & ~(size_t)255); return p; };
    auto alloci = [&](size_t n) { int*   p = (int*)(base + off);   off += ((n * 4 + 255) & ~(size_t)255); return p; };
    auto allocu = [&](size_t n) { ushort_t* p = (ushort_t*)(base + off); off += ((n * 2 + 255) & ~(size_t)255); return p; };

    float* cx   = allocf((size_t)Nn * 512);
    float* hbuf = allocf((size_t)Nn * 1024);
    float* hn   = allocf((size_t)Nn * 1024);
    float* xl   = allocf((size_t)Nn * 1024);
    float* xr   = allocf((size_t)Nn * 1024);
    float* mb   = allocf((size_t)Nn * 2);
    float* den  = allocf((size_t)Nn * 2);
    float* elog = allocf((size_t)E2 * 2);
    float* ep   = allocf((size_t)E2 * 2);
    float* gbuf = allocf((size_t)Bn * 512);
    float* o1   = allocf((size_t)Bn * 384);
    float* o2   = allocf((size_t)Bn * 256);
    int* src2 = alloci(E2);
    int* dst2 = alloci(E2);

    auto pack = [&](int idx, int K, int Nc) {
        ushort_t* p = allocu((size_t)K * Nc);
        int total = K * Nc;
        k_pack_w<<<(total + 255) / 256, 256, 0, stream>>>((const float*)d_in[idx], p, K, Nc);
        return p;
    };

    ushort_t* p_xtw1 = pack(3, 32, 256);
    ushort_t* p_xtw2 = pack(5, 256, 256);
    const int wlIdx[5] = {7, 11, 15, 19, 23};
    const int Kin[5]   = {512, 1024, 1024, 1024, 1024};
    const int Nout[5]  = {1024, 1024, 1024, 1024, 512};
    const int Hh[5]    = {2, 2, 2, 2, 1};
    ushort_t* p_wl[5]; ushort_t* p_wr[5];
    for (int l = 0; l < 5; ++l) {
        p_wl[l] = pack(wlIdx[l],     Kin[l], Nout[l]);
        p_wr[l] = pack(wlIdx[l] + 1, Kin[l], Nout[l]);
    }
    ushort_t* p_m1 = pack(30, 512, 384);
    ushort_t* p_m2 = pack(32, 384, 256);
    ushort_t* p_m3 = pack(34, 256, 128);

    // grid.x * 4 waves * NT == Nc/16 exactly (Nc multiple of 256 -> NT=4,
    // otherwise Nc multiple of 128 -> NT=2); no tail guards in the kernel.
    auto gemm = [&](const float* A, const ushort_t* Wp, const float* bias, float* C,
                    int M, int Nc, int K, int act) {
        int nTiles = Nc / 16;
        if (nTiles % 16 == 0) {
            dim3 grid(nTiles / 16, M / 16);
            k_gemm_bf16<4><<<grid, 128, 0, stream>>>(A, Wp, bias, C, Nc, K, act);
        } else {
            dim3 grid(nTiles / 8, M / 16);
            k_gemm_bf16<2><<<grid, 128, 0, stream>>>(A, Wp, bias, C, Nc, K, act);
        }
    };

    // edges + self loops
    k_build_edges<<<(E2 + 255) / 256, 256, 0, stream>>>(ei, src2, dst2, Eo, Nn);

    // x_trans: Linear -> LeakyReLU -> Linear, fold qubit-pairs into [N, 512]
    float* tmp = xl;   // reuse before GAT projections
    gemm(x,   p_xtw1, (const float*)d_in[4], tmp, 2 * Nn, 256, 32,  1);
    gemm(tmp, p_xtw2, (const float*)d_in[6], cx,  2 * Nn, 256, 256, 0);

    const int attIdx[5] = {9, 13, 17, 21, 25};
    const int bIdx[5]   = {10, 14, 18, 22, 26};
    const float* inptr = cx;
    for (int l = 0; l < 5; ++l) {
        const int H = Hh[l], No = Nout[l];
        gemm(inptr, p_wl[l], nullptr, xl, Nn, No, Kin[l], 0);
        gemm(inptr, p_wr[l], nullptr, xr, Nn, No, Kin[l], 0);
        k_fill<<<(Nn * H + 255) / 256, 256, 0, stream>>>(mb, -1e30f, Nn * H);
        k_fill<<<(Nn * H + 255) / 256, 256, 0, stream>>>(den, 0.f, Nn * H);
        k_fill<<<(Nn * No + 255) / 256, 256, 0, stream>>>(hn, 0.f, Nn * No);
        int waves = E2 * H;
        k_edge_logit<<<(waves * 32 + 255) / 256, 256, 0, stream>>>(
            xl, xr, (const float*)d_in[attIdx[l]], src2, dst2, elog, mb, E2, H, No);
        k_edge_alpha<<<(E2 * H + 255) / 256, 256, 0, stream>>>(elog, mb, dst2, ep, den, E2, H);
        k_edge_scatter<<<E2, 256, 0, stream>>>(xl, ep, den, src2, dst2, hn, H, No);
        k_bias_lrelu<<<(Nn * No + 255) / 256, 256, 0, stream>>>(
            hn, (const float*)d_in[bIdx[l]], hbuf, Nn * No, No);
        inptr = hbuf;
    }

    // residual + graph LayerNorm + softmax aggregation
    k_add<<<(Nn * 512 + 255) / 256, 256, 0, stream>>>(hbuf, cx, Nn * 512);
    k_layernorm<<<Bn, 256, 0, stream>>>(hbuf, (const float*)d_in[27], (const float*)d_in[28], npg);
    k_aggr<<<Bn, 256, 0, stream>>>(hbuf, (const float*)d_in[29], gbuf, npg);

    // head MLP 512 -> 384 -> 256 -> 128
    gemm(gbuf, p_m1, (const float*)d_in[31], o1, Bn, 384, 512, 1);
    gemm(o1,   p_m2, (const float*)d_in[33], o2, Bn, 256, 384, 1);
    gemm(o2,   p_m3, (const float*)d_in[35], (float*)d_out, Bn, 128, 256, 0);
}
